// snNet_63522566307849
// MI455X (gfx1250) — matrix operand order repaired and checked by reading, not compiled, not run on hardware
//
#include <hip/hip_runtime.h>
#include <hip/hip_bf16.h>

// ---------------------------------------------------------------------------
// SNN (2-layer LIF) forward, T=100 steps, B=1024, 784 -> 1000 -> 10.
// Layer-1 GEMM: v_wmma_f32_16x16x32_f16, 32x64 tile/wave, LDS double-buffered.
// ---------------------------------------------------------------------------

#define T_STEPS 100
#define BATCH   1024
#define N_IN    784
#define N_HID   1000
#define N_OUT   10

#define KP1 800    // N_IN padded to multiple of 32
#define NP1 1024   // N_HID padded to multiple of 128
#define KP2 1024   // layer-2 K = NP1
#define NP2 16     // N_OUT padded to one 16-wide tile

#define BETA 0.9f
#define THR  1.0f

typedef __attribute__((ext_vector_type(16))) _Float16 v16h;
typedef __attribute__((ext_vector_type(8)))  float    v8f;

union Frag16 { v16h h; uint4 u[2]; };

// --------------------------- prep kernels ----------------------------------

__global__ void k_prep_w1(const float* __restrict__ W1, _Float16* __restrict__ w1h) {
    int i = blockIdx.x * blockDim.x + threadIdx.x;      // over NP1*KP1
    if (i >= NP1 * KP1) return;
    int n = i / KP1, k = i - n * KP1;
    float v = (n < N_HID && k < N_IN) ? W1[n * N_IN + k] : 0.0f;
    w1h[i] = (_Float16)v;
}

__global__ void k_prep_w2(const float* __restrict__ W2, _Float16* __restrict__ w2h) {
    int i = blockIdx.x * blockDim.x + threadIdx.x;      // over NP2*KP2
    if (i >= NP2 * KP2) return;
    int n = i >> 10, k = i & (KP2 - 1);
    float v = (n < N_OUT && k < N_HID) ? W2[n * N_HID + k] : 0.0f;
    w2h[i] = (_Float16)v;
}

__global__ void k_zero_state(float* __restrict__ mem1, float* __restrict__ mem2) {
    int i = blockIdx.x * blockDim.x + threadIdx.x;
    if (i < BATCH * NP1) mem1[i] = 0.0f;
    if (i < BATCH * NP2) mem2[i] = 0.0f;
}

// --------------------- layer 1: GEMM + LIF (fused) -------------------------
// grid: (NP1/128, BATCH/64), block: 128 threads (4 waves).
// WG tile 64(M) x 128(N); wave grid 2x2; wave tile 32(M) x 64(N) = 2x4 wmma.
// LDS double-buffered; waves_per_eu(1) lifts the VGPR cap so staging loads
// clause-batch instead of serializing through one recycled register quad.

__global__ __launch_bounds__(128) __attribute__((amdgpu_waves_per_eu(1)))
void k_layer1(
    const float*    __restrict__ xt,     // [BATCH][N_IN], this timestep
    const _Float16* __restrict__ w1h,    // [NP1][KP1]  (B as [n][k])
    const float*    __restrict__ b1,     // [N_HID]
    float*          __restrict__ mem1,   // [BATCH][NP1] state
    _Float16*       __restrict__ spk1h)  // [BATCH][KP2] spikes (f16)
{
    __shared__ __align__(16) _Float16 lA[2][64][40];    // [buf][m][k], padded
    __shared__ __align__(16) _Float16 lB[2][128][40];   // [buf][n][k], padded

    const int tid  = threadIdx.x;
    const int lane = tid & 31;
    const int wave = tid >> 5;
    const int nBase = blockIdx.x * 128;
    const int mBase = blockIdx.y * 64;
    const int mW = (wave >> 1) * 32;     // wave M offset in WG tile
    const int nW = (wave & 1) * 64;      // wave N offset in WG tile

    v8f c[2][4] = {};

    // A loader: 2 threads per row, 16 halves each. B loader: 1 thread per row.
    const int alr = tid >> 1;            // 0..63
    const int alk = (tid & 1) * 16;      // 0 or 16

    // per-lane fragment offsets (ISA 7.12.2 layouts)
    const int ar = lane & 15;            // A row / B column within 16-tile
    const int ak = (lane >> 4) * 8;      // A: half-wave K split (8+8 halves)
    const int bk = (lane >> 4) * 8;      // B: half-wave K split (8+8 halves)

    const float* aRow = xt + (size_t)(mBase + alr) * N_IN;
    const uint4* bRow = (const uint4*)(w1h + (size_t)(nBase + tid) * KP1);

    // ---- fused fetch+stage: distinct locals -> loads issue as one clause
    auto stageA16 = [&](int buf, const float4* src) {
        float4 f0 = src[0], f1 = src[1], f2 = src[2], f3 = src[3];
        _Float16 tmp[16];
        tmp[0]  = (_Float16)f0.x; tmp[1]  = (_Float16)f0.y;
        tmp[2]  = (_Float16)f0.z; tmp[3]  = (_Float16)f0.w;
        tmp[4]  = (_Float16)f1.x; tmp[5]  = (_Float16)f1.y;
        tmp[6]  = (_Float16)f1.z; tmp[7]  = (_Float16)f1.w;
        tmp[8]  = (_Float16)f2.x; tmp[9]  = (_Float16)f2.y;
        tmp[10] = (_Float16)f2.z; tmp[11] = (_Float16)f2.w;
        tmp[12] = (_Float16)f3.x; tmp[13] = (_Float16)f3.y;
        tmp[14] = (_Float16)f3.z; tmp[15] = (_Float16)f3.w;
        *(uint4*)&lA[buf][alr][alk]     = *(const uint4*)&tmp[0];
        *(uint4*)&lA[buf][alr][alk + 8] = *(const uint4*)&tmp[8];
    };
    auto stageAzero = [&](int buf) {
        uint4 z = make_uint4(0u, 0u, 0u, 0u);
        *(uint4*)&lA[buf][alr][alk]     = z;
        *(uint4*)&lA[buf][alr][alk + 8] = z;
    };
    auto stageB = [&](int buf, int kb) {
        const uint4* src = bRow + (kb >> 3);   // kb halves -> kb/8 uint4
        uint4 r0 = src[0], r1 = src[1], r2 = src[2], r3 = src[3];
        *(uint4*)&lB[buf][tid][0]  = r0;
        *(uint4*)&lB[buf][tid][8]  = r1;
        *(uint4*)&lB[buf][tid][16] = r2;
        *(uint4*)&lB[buf][tid][24] = r3;
    };
    auto fetchstage_full = [&](int buf, int kb) {
        stageA16(buf, (const float4*)(aRow + kb + alk));
        stageB(buf, kb);
    };
    auto fetchstage_tail = [&](int buf, int kb) {   // only kb == 768
        if (kb + alk < N_IN) stageA16(buf, (const float4*)(aRow + kb + alk));
        else                 stageAzero(buf);
        stageB(buf, kb);
    };

    // ---- compute: all frags loaded up front -> staggered partial dscnt waits
    auto compute = [&](int buf) {
        Frag16 a[2], b[4];
#pragma unroll
        for (int i = 0; i < 2; ++i) {
            const _Float16* p = &lA[buf][mW + i * 16 + ar][ak];
            a[i].u[0] = *(const uint4*)p;
            a[i].u[1] = *(const uint4*)(p + 16);
        }
#pragma unroll
        for (int j = 0; j < 4; ++j) {
            const _Float16* p = &lB[buf][nW + j * 16 + ar][bk];
            b[j].u[0] = *(const uint4*)p;
            b[j].u[1] = *(const uint4*)(p + 16);
        }
#pragma unroll
        for (int j = 0; j < 4; ++j) {
            c[0][j] = __builtin_amdgcn_wmma_f32_16x16x32_f16(
                false, a[0].h, false, b[j].h, (short)0, c[0][j], false, false);
            c[1][j] = __builtin_amdgcn_wmma_f32_16x16x32_f16(
                false, a[1].h, false, b[j].h, (short)0, c[1][j], false, false);
        }
    };

    // ---- prologue: fill buffer 0 with tile kb=0 (always in-bounds)
    fetchstage_full(0, 0);
    __syncthreads();

    int cur = 0;
    // main loop: tiles 32..736 staged, all fully in-bounds (kb+31 < 784)
    for (int kb = 32; kb < KP1 - 32; kb += 32) {
        compute(cur);                    // 8 wmma from LDS[cur]
        fetchstage_full(cur ^ 1, kb);    // stage next tile
        __syncthreads();
        cur ^= 1;
    }
    // peeled: stage final (partially padded) tile kb = 768
    compute(cur);
    fetchstage_tail(cur ^ 1, KP1 - 32);
    __syncthreads();
    cur ^= 1;
    compute(cur);

    // ---- LIF epilogue: mem1 = beta*mem1 + (cur1 + b1) - reset; spk1 f16
    const int col    = lane & 15;
    const int rowOff = (lane >> 4) * 8;
#pragma unroll
    for (int i = 0; i < 2; ++i) {
#pragma unroll
        for (int j = 0; j < 4; ++j) {
            const int n = nBase + nW + j * 16 + col;
#pragma unroll
            for (int vg = 0; vg < 8; ++vg) {
                const int m = mBase + mW + i * 16 + rowOff + vg;
                const size_t idx = (size_t)m * NP1 + n;
                if (n < N_HID) {
                    float cur1  = c[i][j][vg] + b1[n];
                    float old   = mem1[idx];
                    float reset = (old > THR) ? THR : 0.0f;
                    float nm    = BETA * old + cur1 - reset;
                    mem1[idx]  = nm;
                    spk1h[idx] = (_Float16)((nm > THR) ? 1.0f : 0.0f);
                } else {
                    spk1h[idx] = (_Float16)0.0f;   // keep layer-2 K-padding clean
                }
            }
        }
    }
}

// --------------------- layer 2: GEMM + LIF (fused) -------------------------
// grid: 8 blocks x 256 threads (8 waves); each wave owns one 16-row M tile,
// single 16-wide N tile (only n < 10 is real), K = 1024 in 32 steps.
// Two interleaved accumulators break the wmma RAW dependency chain.

__global__ __launch_bounds__(256) void k_layer2(
    const _Float16* __restrict__ spk1h,  // [BATCH][KP2]
    const _Float16* __restrict__ w2h,    // [NP2][KP2]
    const float*    __restrict__ b2,     // [N_OUT]
    float*          __restrict__ mem2,   // [BATCH][NP2] state
    float*          __restrict__ out_spk,// [BATCH][N_OUT] slice of d_out
    float*          __restrict__ out_mem)// [BATCH][N_OUT] slice of d_out
{
    const int tid  = threadIdx.x;
    const int lane = tid & 31;
    const int wave = tid >> 5;
    const int mBase = (blockIdx.x * 8 + wave) * 16;

    const int ar = lane & 15;
    const int ak = (lane >> 4) * 8;
    const int bk = (lane >> 4) * 8;

    const _Float16* paBase = spk1h + (size_t)(mBase + ar) * KP2;
    const _Float16* pbBase = w2h + (size_t)ar * KP2;

    v8f c0 = {}, c1 = {};
#pragma unroll 2
    for (int kb = 0; kb < KP2; kb += 64) {
        Frag16 a0, b0, a1, b1f;
        const _Float16* pa = paBase + kb + ak;
        const _Float16* pb = pbBase + kb + bk;
        a0.u[0]  = *(const uint4*)pa;
        a0.u[1]  = *(const uint4*)(pa + 16);
        b0.u[0]  = *(const uint4*)pb;
        b0.u[1]  = *(const uint4*)(pb + 16);
        a1.u[0]  = *(const uint4*)(pa + 32);
        a1.u[1]  = *(const uint4*)(pa + 48);
        b1f.u[0] = *(const uint4*)(pb + 32);
        b1f.u[1] = *(const uint4*)(pb + 48);
        c0 = __builtin_amdgcn_wmma_f32_16x16x32_f16(
                 false, a0.h, false, b0.h, (short)0, c0, false, false);
        c1 = __builtin_amdgcn_wmma_f32_16x16x32_f16(
                 false, a1.h, false, b1f.h, (short)0, c1, false, false);
    }
    v8f c = c0 + c1;

    const int n = lane & 15;
    if (n < N_OUT) {
#pragma unroll
        for (int vg = 0; vg < 8; ++vg) {
            const int m = mBase + (lane >> 4) * 8 + vg;
            const size_t sidx = (size_t)m * NP2 + n;
            float cur   = c[vg] + b2[n];
            float old   = mem2[sidx];
            float reset = (old > THR) ? THR : 0.0f;
            float nm    = BETA * old + cur - reset;
            mem2[sidx] = nm;
            out_spk[(size_t)m * N_OUT + n] = (nm > THR) ? 1.0f : 0.0f;
            out_mem[(size_t)m * N_OUT + n] = nm;
        }
    }
}

// ------------------------------- launcher ----------------------------------

extern "C" void kernel_launch(void* const* d_in, const int* in_sizes, int n_in,
                              void* d_out, int out_size, void* d_ws, size_t ws_size,
                              hipStream_t stream) {
    const float* x  = (const float*)d_in[0];   // [T][B][784]
    const float* W1 = (const float*)d_in[1];   // [1000][784]
    const float* b1 = (const float*)d_in[2];   // [1000]
    const float* W2 = (const float*)d_in[3];   // [10][1000]
    const float* b2 = (const float*)d_in[4];   // [10]
    float* out = (float*)d_out;                // spk_rec | mem_rec, each T*B*10

    // workspace carve-up (all 256B aligned)
    char* ws = (char*)d_ws;
    _Float16* w1h   = (_Float16*)(ws + 0);              // NP1*KP1*2 = 1,638,400
    _Float16* w2h   = (_Float16*)(ws + 1638400);        // NP2*KP2*2 =    32,768
    float*    mem1  = (float*)   (ws + 1671168);        // B*NP1*4   = 4,194,304
    _Float16* spk1h = (_Float16*)(ws + 5865472);        // B*KP2*2   = 2,097,152
    float*    mem2  = (float*)   (ws + 7962624);        // B*NP2*4   =    65,536
    // total 8,028,160 bytes

    k_prep_w1<<<(NP1 * KP1 + 255) / 256, 256, 0, stream>>>(W1, w1h);
    k_prep_w2<<<(NP2 * KP2 + 255) / 256, 256, 0, stream>>>(W2, w2h);
    k_zero_state<<<(BATCH * NP1 + 255) / 256, 256, 0, stream>>>(mem1, mem2);

    const dim3 g1(NP1 / 128, BATCH / 64);   // 8 x 16 workgroups
    const size_t xStep   = (size_t)BATCH * N_IN;
    const size_t outStep = (size_t)BATCH * N_OUT;
    const size_t memOfs  = (size_t)T_STEPS * BATCH * N_OUT;

    for (int t = 0; t < T_STEPS; ++t) {
        k_layer1<<<g1, 128, 0, stream>>>(x + (size_t)t * xStep, w1h, b1, mem1, spk1h);
        k_layer2<<<8, 256, 0, stream>>>(spk1h, w2h, b2, mem2,
                                        out + (size_t)t * outStep,
                                        out + memOfs + (size_t)t * outStep);
    }
}
